// PointTransformerSeg_420906795559
// MI455X (gfx1250) — compile-verified, matrix operand order checked by reading, not run on hardware
//
#include <hip/hip_runtime.h>
#include <hip/hip_bf16.h>
#include <math.h>

// ---------------------------------------------------------------------------
// PointTransformer forward for MI455X (gfx1250, wave32, WMMA).
// GEMMs: LDS-staged 64x64 block tiles, 8 waves, 2x v_wmma_f32_16x16x32_f16
// per wave per K-slab, fragments read as ds_load_b128, staging via 16B
// coalesced global loads on a block-uniform fast path. BN uses live batch
// statistics (reduce + fused affine/ReLU/residual). kNN is templated on K so
// the top-k lives in VGPRs; attention gathers are fused into the consumers.
// ---------------------------------------------------------------------------

typedef _Float16 f16;
typedef __attribute__((ext_vector_type(16))) _Float16 v16h;
typedef __attribute__((ext_vector_type(8)))  _Float16 v8h;
typedef __attribute__((ext_vector_type(4)))  _Float16 v4h;
typedef __attribute__((ext_vector_type(8)))  float    v8f;
typedef __attribute__((ext_vector_type(4)))  float    v4f;

#define EPS_BN 1e-5f

static inline int ceil_div(int a, int b) { return (a + b - 1) / b; }

// ---------------------------------------------------------------------------
// WMMA GEMM: Y[M,N] = X[M,K] * W[K,N] (+ bias[N]).
// Block tile 64(M) x 64(N), K swept in 32-slabs. 8 waves; wave (wm, wn) owns
// rows [wm*16,+16) and cols [wn*16,+32) -> two 16x16 WMMA accumulators.
// LDS: A tile row-major [64][32] halfs, B tile transposed [64 n][32 k] halfs,
// rows padded to 40 halfs (80B: 16B-aligned b128 reads, skewed banks).
// ---------------------------------------------------------------------------
#define GBM 64
#define GBN 64
#define GBK 32
#define LDH 40

__global__ void k_gemm_wmma(const float* __restrict__ X, const float* __restrict__ W,
                            const float* __restrict__ bias, float* __restrict__ Y,
                            int M, int K, int N) {
  __shared__ __align__(16) f16 sA[GBM * LDH];
  __shared__ __align__(16) f16 sB[GBN * LDH];

  const int tid  = threadIdx.x;
  const int lane = tid & 31;
  const int wave = tid >> 5;
  const int tiles_n = (N + GBN - 1) / GBN;
  const int bm = (blockIdx.x / tiles_n) * GBM;
  const int bn = (blockIdx.x % tiles_n) * GBN;

  const int wm = wave >> 1;        // 0..3 : 16-row slab
  const int wn = (wave & 1) * 2;   // 0 or 2 : pair of 16-col subtiles
  const int l15 = lane & 15;
  const int kb  = (lane >> 4) << 3;

  const bool fullM = (bm + GBM <= M);
  const bool fullN = (bn + GBN <= N);

  v8f acc0 = {};
  v8f acc1 = {};

  for (int k0 = 0; k0 < K; k0 += GBK) {
    const bool fullK = (k0 + GBK <= K);
    if (k0 + GBK < K) {  // speculative prefetch of next K-slab
      __builtin_prefetch(&X[(long)(bm + (tid >> 3)) * K + k0 + GBK], 0, 1);
      __builtin_prefetch(&W[(long)(k0 + GBK) * N + bn + (tid & 63)], 0, 1);
    }

    // ---- stage A tile: 64 rows x 32 K --------------------------------------
    if (fullM && fullK) {
#pragma unroll
      for (int t = 0; t < 2; ++t) {
        const int e = (tid + t * 256) << 2;   // 0..2044, step 4
        const int r = e >> 5, k = e & 31;
        v4f v;
        __builtin_memcpy(&v, &X[(long)(bm + r) * K + k0 + k], 16);
        v4h h; h[0] = (f16)v[0]; h[1] = (f16)v[1]; h[2] = (f16)v[2]; h[3] = (f16)v[3];
        *(v4h*)&sA[r * LDH + k] = h;
      }
    } else {
#pragma unroll
      for (int t = 0; t < 8; ++t) {
        const int e = tid + t * 256;          // 0..2047
        const int r = e >> 5, k = e & 31;
        const int gr = bm + r, gk = k0 + k;
        const int cr = gr < M ? gr : M - 1;
        const int ck = gk < K ? gk : K - 1;
        const float v = X[(long)cr * K + ck];
        sA[r * LDH + k] = (f16)((gr < M && gk < K) ? v : 0.f);
      }
    }

    // ---- stage B tile transposed: [n][k] -----------------------------------
    if (fullN && fullK) {
#pragma unroll
      for (int t = 0; t < 2; ++t) {
        const int e = (tid + t * 256) << 2;   // 0..2044, step 4
        const int k = e >> 6, n = e & 63;
        v4f v;
        __builtin_memcpy(&v, &W[(long)(k0 + k) * N + bn + n], 16);
        sB[(n + 0) * LDH + k] = (f16)v[0];
        sB[(n + 1) * LDH + k] = (f16)v[1];
        sB[(n + 2) * LDH + k] = (f16)v[2];
        sB[(n + 3) * LDH + k] = (f16)v[3];
      }
    } else {
#pragma unroll
      for (int t = 0; t < 8; ++t) {
        const int e = tid + t * 256;          // 0..2047
        const int k = e >> 6, n = e & 63;
        const int gk = k0 + k, gn = bn + n;
        const int ck = gk < K ? gk : K - 1;
        const int cn = gn < N ? gn : N - 1;
        const float v = W[(long)ck * N + cn];
        sB[n * LDH + k] = (f16)((gk < K && gn < N) ? v : 0.f);
      }
    }
    __syncthreads();

    // ---- fragments: 2x ds_load_b128 each -----------------------------------
    const f16* pa  = &sA[(wm * 16 + l15) * LDH + kb];
    const f16* pb0 = &sB[(wn * 16 + l15) * LDH + kb];
    const f16* pb1 = &sB[((wn + 1) * 16 + l15) * LDH + kb];
    const v8h alo = *(const v8h*)pa;
    const v8h ahi = *(const v8h*)(pa + 16);
    const v8h b0lo = *(const v8h*)pb0;
    const v8h b0hi = *(const v8h*)(pb0 + 16);
    const v8h b1lo = *(const v8h*)pb1;
    const v8h b1hi = *(const v8h*)(pb1 + 16);
    v16h a, b0, b1;
#pragma unroll
    for (int j = 0; j < 8; ++j) {
      a[j] = alo[j];  a[j + 8] = ahi[j];
      b0[j] = b0lo[j]; b0[j + 8] = b0hi[j];
      b1[j] = b1lo[j]; b1[j + 8] = b1hi[j];
    }
    acc0 = __builtin_amdgcn_wmma_f32_16x16x32_f16(false, a, false, b0,
                                                  (short)0, acc0, false, false);
    acc1 = __builtin_amdgcn_wmma_f32_16x16x32_f16(false, a, false, b1,
                                                  (short)0, acc1, false, false);
    __syncthreads();
  }

  // ---- epilogue: D tile row = r + (lane/16)*8, col = lane%16 ---------------
  const int rowbase = bm + wm * 16 + ((lane >> 4) << 3);
  const int oc0 = bn + wn * 16 + l15;
  const int oc1 = oc0 + 16;
  const float bia0 = (bias && oc0 < N) ? bias[oc0] : 0.f;
  const float bia1 = (bias && oc1 < N) ? bias[oc1] : 0.f;
#pragma unroll
  for (int r = 0; r < 8; ++r) {
    const int orow = rowbase + r;
    if (orow < M) {
      if (oc0 < N) Y[(long)orow * N + oc0] = acc0[r] + bia0;
      if (oc1 < N) Y[(long)orow * N + oc1] = acc1[r] + bia1;
    }
  }
}

// ---------------------------------------------------------------------------
// BN statistics: one block per channel, reduce sum/sumsq over rows.
// ---------------------------------------------------------------------------
__global__ void k_bnstats(const float* __restrict__ X, float* __restrict__ mean,
                          float* __restrict__ var, int rows, int C) {
  __shared__ float ssum[256];
  __shared__ float ssq[256];
  const int c = blockIdx.x;
  float s = 0.f, q = 0.f;
  for (int r = threadIdx.x; r < rows; r += blockDim.x) {
    float v = X[(long)r * C + c];
    s += v; q += v * v;
  }
  ssum[threadIdx.x] = s; ssq[threadIdx.x] = q;
  __syncthreads();
  for (int st = 128; st > 0; st >>= 1) {
    if (threadIdx.x < st) {
      ssum[threadIdx.x] += ssum[threadIdx.x + st];
      ssq[threadIdx.x]  += ssq[threadIdx.x + st];
    }
    __syncthreads();
  }
  if (threadIdx.x == 0) {
    float m = ssum[0] / (float)rows;
    mean[c] = m;
    var[c]  = ssq[0] / (float)rows - m * m;
  }
}

// Fused BN-affine (+ optional residual, + optional ReLU).
__global__ void k_bn_apply(const float* __restrict__ X, const float* __restrict__ mean,
                           const float* __restrict__ var, const float* __restrict__ g,
                           const float* __restrict__ b, const float* __restrict__ resid,
                           float* __restrict__ Y, int total, int C, int relu) {
  int tid = blockIdx.x * blockDim.x + threadIdx.x;
  if (tid >= total) return;
  int c = tid % C;
  float v = (X[tid] - mean[c]) * rsqrtf(var[c] + EPS_BN) * g[c] + b[c];
  if (resid) v += resid[tid];
  if (relu) v = fmaxf(v, 0.f);
  Y[tid] = v;
}

// ---------------------------------------------------------------------------
// kNN, templated on K: top-k entirely in VGPRs (static indices), LDS tiles of
// 256 reference points. Insert = guarded fully-unrolled compare-swap chain.
// ---------------------------------------------------------------------------
#define KNN_TILE 256
template <int KT>
__global__ void k_knn_t(const float* __restrict__ Q, const float* __restrict__ R,
                        int* __restrict__ oidx, float* __restrict__ odist,
                        int NQ, int NR) {
  __shared__ float sx[KNN_TILE], sy[KNN_TILE], sz[KNN_TILE];
  const int b = blockIdx.y;
  const int i = blockIdx.x * blockDim.x + threadIdx.x;
  const bool valid = (i < NQ);
  float qx = 0.f, qy = 0.f, qz = 0.f;
  if (valid) {
    qx = Q[((long)b * NQ + i) * 3 + 0];
    qy = Q[((long)b * NQ + i) * 3 + 1];
    qz = Q[((long)b * NQ + i) * 3 + 2];
  }
  float bd[KT]; int bi[KT];
#pragma unroll
  for (int t = 0; t < KT; ++t) { bd[t] = 3.0e38f; bi[t] = 0; }

  for (int base = 0; base < NR; base += KNN_TILE) {
    const int nt = (NR - base < KNN_TILE) ? (NR - base) : KNN_TILE;
    __syncthreads();
    for (int j = threadIdx.x; j < nt; j += blockDim.x) {
      sx[j] = R[((long)b * NR + base + j) * 3 + 0];
      sy[j] = R[((long)b * NR + base + j) * 3 + 1];
      sz[j] = R[((long)b * NR + base + j) * 3 + 2];
    }
    __syncthreads();
    if (valid) {
      for (int j = 0; j < nt; ++j) {
        float dx = qx - sx[j], dy = qy - sy[j], dz = qz - sz[j];
        float d = dx * dx + dy * dy + dz * dz;
        if (d < bd[KT - 1]) {               // rare after warm-up
          bd[KT - 1] = d; bi[KT - 1] = base + j;
#pragma unroll
          for (int t = KT - 1; t > 0; --t) {  // bubble-up, static indices
            const bool sw = bd[t] < bd[t - 1];
            const float d0 = sw ? bd[t] : bd[t - 1];
            const float d1 = sw ? bd[t - 1] : bd[t];
            const int   i0 = sw ? bi[t] : bi[t - 1];
            const int   i1 = sw ? bi[t - 1] : bi[t];
            bd[t - 1] = d0; bd[t] = d1; bi[t - 1] = i0; bi[t] = i1;
          }
        }
      }
    }
  }
  if (valid) {
#pragma unroll
    for (int t = 0; t < KT; ++t) {
      oidx[((long)b * NQ + i) * KT + t]  = bi[t];
      odist[((long)b * NQ + i) * KT + t] = sqrtf(fmaxf(bd[t], 0.f));
    }
  }
}

// ---------------------------------------------------------------------------
// Farthest point sampling: one workgroup per batch; distance field in LDS.
// ---------------------------------------------------------------------------
#define FPS_THREADS 512
__global__ void k_fps(const float* __restrict__ P, int* __restrict__ sidx, int n, int m) {
  __shared__ float sdist[8192];
  __shared__ float rmax[FPS_THREADS];
  __shared__ int   rarg[FPS_THREADS];
  const int b = blockIdx.x;
  const float* p = P + (long)b * n * 3;
  for (int i = threadIdx.x; i < n; i += blockDim.x) sdist[i] = 1.0e10f;
  __syncthreads();
  int last = 0;
  for (int it = 0; it < m; ++it) {
    if (threadIdx.x == 0) sidx[(long)b * m + it] = last;
    const float lx = p[last * 3 + 0], ly = p[last * 3 + 1], lz = p[last * 3 + 2];
    float best = -1.f; int bestj = 0;
    for (int i = threadIdx.x; i < n; i += blockDim.x) {
      float dx = p[i * 3 + 0] - lx, dy = p[i * 3 + 1] - ly, dz = p[i * 3 + 2] - lz;
      float d = dx * dx + dy * dy + dz * dz;
      float cur = sdist[i];
      if (d < cur) { cur = d; sdist[i] = d; }
      if (cur > best) { best = cur; bestj = i; }  // first-max (ascending i)
    }
    rmax[threadIdx.x] = best; rarg[threadIdx.x] = bestj;
    __syncthreads();
    for (int st = FPS_THREADS / 2; st > 0; st >>= 1) {
      if (threadIdx.x < st) {
        float ov = rmax[threadIdx.x + st]; int oi = rarg[threadIdx.x + st];
        float mv = rmax[threadIdx.x];      int mi = rarg[threadIdx.x];
        if (ov > mv || (ov == mv && oi < mi)) { rmax[threadIdx.x] = ov; rarg[threadIdx.x] = oi; }
      }
      __syncthreads();
    }
    last = rarg[0];
    __syncthreads();
  }
}

// ---------------------------------------------------------------------------
// Elementwise / gather / pooling kernels.
// ---------------------------------------------------------------------------
__global__ void k_transpose_in(const float* __restrict__ data, float* __restrict__ x0,
                               float* __restrict__ p0, int Bn, int Cin, int Np) {
  int tid = blockIdx.x * blockDim.x + threadIdx.x;
  int total = Bn * Np * Cin;
  if (tid >= total) return;
  int c = tid % Cin; int i = (tid / Cin) % Np; int b = tid / (Cin * Np);
  float v = data[((long)b * Cin + c) * Np + i];
  x0[((long)b * Np + i) * Cin + c] = v;
  if (c < 3) p0[((long)b * Np + i) * 3 + c] = v;
}

__global__ void k_transpose_out(const float* __restrict__ Z, float* __restrict__ out,
                                int Bn, int Np, int C) {
  int tid = blockIdx.x * blockDim.x + threadIdx.x;
  int total = Bn * Np * C;
  if (tid >= total) return;
  int c = tid % C; int i = (tid / C) % Np; int b = tid / (C * Np);
  out[((long)b * C + c) * Np + i] = Z[((long)b * Np + i) * C + c];
}

__global__ void k_gather_feat(const float* __restrict__ X, const int* __restrict__ idx,
                              float* __restrict__ out, int Bn, int M, int K, int C, int NR) {
  int tid = blockIdx.x * blockDim.x + threadIdx.x;
  int total = Bn * M * K * C;
  if (tid >= total) return;
  int c = tid % C; int kk = (tid / C) % K; int i = (tid / (C * K)) % M; int b = tid / (C * K * M);
  int j = idx[((long)b * M + i) * K + kk];
  out[tid] = X[((long)b * NR + j) * C + c];
}

// out[b,i,k,:] = P[b, idx[b,i,k], :] - Q[b,i,:]
__global__ void k_gather_sub3(const float* __restrict__ P, const int* __restrict__ idx,
                              const float* __restrict__ Q, float* __restrict__ out,
                              int Bn, int M, int K, int NR) {
  int tid = blockIdx.x * blockDim.x + threadIdx.x;
  int total = Bn * M * K * 3;
  if (tid >= total) return;
  int c = tid % 3; int kk = (tid / 3) % K; int i = (tid / (3 * K)) % M; int b = tid / (3 * K * M);
  int j = idx[((long)b * M + i) * K + kk];
  out[tid] = P[((long)b * NR + j) * 3 + c] - Q[((long)b * M + i) * 3 + c];
}

__global__ void k_gather_pts(const float* __restrict__ P, const int* __restrict__ sidx,
                             float* __restrict__ out, int Bn, int M, int NR) {
  int tid = blockIdx.x * blockDim.x + threadIdx.x;
  int total = Bn * M * 3;
  if (tid >= total) return;
  int c = tid % 3; int i = (tid / 3) % M; int b = tid / (3 * M);
  int j = sidx[(long)b * M + i];
  out[tid] = P[((long)b * NR + j) * 3 + c];
}

__global__ void k_concat(const float* __restrict__ A, const float* __restrict__ Bm,
                         float* __restrict__ out, int rows, int Ca, int Cb) {
  int tid = blockIdx.x * blockDim.x + threadIdx.x;
  int total = rows * (Ca + Cb);
  if (tid >= total) return;
  int c = tid % (Ca + Cb); int r = tid / (Ca + Cb);
  out[tid] = (c < Ca) ? A[(long)r * Ca + c] : Bm[(long)r * Cb + (c - Ca)];
}

// xc[b,i,:] = concat(x[b,i,:], g[b,:])
__global__ void k_concat_bcast(const float* __restrict__ X, const float* __restrict__ G,
                               float* __restrict__ out, int Bn, int Np, int C) {
  int tid = blockIdx.x * blockDim.x + threadIdx.x;
  int total = Bn * Np * 2 * C;
  if (tid >= total) return;
  int c = tid % (2 * C); int i = (tid / (2 * C)) % Np; int b = tid / (2 * C * Np);
  out[tid] = (c < C) ? X[((long)b * Np + i) * C + c] : G[(long)b * C + (c - C)];
}

// Fused gather + broadcast-sub + add:
// w[b,i,k,c] = XK[b, idx[b,i,k], c] - XQ[b,i,c] + PE[b,i,k,c]
__global__ void k_qkpe_g(const float* __restrict__ XK, const int* __restrict__ idx,
                         const float* __restrict__ XQ, const float* __restrict__ PE,
                         float* __restrict__ Wt, int Bn, int Np, int K, int C) {
  int tid = blockIdx.x * blockDim.x + threadIdx.x;
  int total = Bn * Np * K * C;
  if (tid >= total) return;
  int c = tid % C; int kk = (tid / C) % K; int i = (tid / (C * K)) % Np;
  int b = tid / (C * K * Np);
  long bn = (long)b * Np + i;
  int j = idx[bn * K + kk];
  Wt[tid] = XK[((long)b * Np + j) * C + c] - XQ[bn * C + c] + PE[tid];
}

// Fused gather + add + grouped-share weighted sum:
// out[b,i,c] = sum_k (XV[b, idx[b,i,k], c] + PE[b,i,k,c]) * Wt[b,i,k,c % Cs]
__global__ void k_attn_agg_g(const float* __restrict__ XV, const int* __restrict__ idx,
                             const float* __restrict__ PE, const float* __restrict__ Wt,
                             float* __restrict__ out, int Bn, int Np, int K, int C, int Cs) {
  int tid = blockIdx.x * blockDim.x + threadIdx.x;
  int total = Bn * Np * C;
  if (tid >= total) return;
  int c = tid % C; int i = (tid / C) % Np; int b = tid / (C * Np);
  long bn = (long)b * Np + i;
  int t = c % Cs;
  float s = 0.f;
  for (int kk = 0; kk < K; ++kk) {
    int j = idx[bn * K + kk];
    float v = XV[((long)b * Np + j) * C + c] + PE[(bn * K + kk) * C + c];
    s += v * Wt[(bn * K + kk) * Cs + t];
  }
  out[tid] = s;
}

__global__ void k_relu(float* __restrict__ X, int total) {
  int tid = blockIdx.x * blockDim.x + threadIdx.x;
  if (tid >= total) return;
  X[tid] = fmaxf(X[tid], 0.f);
}

// softmax over the K axis of W[rows, K, Cs], in place
__global__ void k_softmax_k(float* __restrict__ Wt, int rows, int K, int Cs) {
  int tid = blockIdx.x * blockDim.x + threadIdx.x;
  int total = rows * Cs;
  if (tid >= total) return;
  int c = tid % Cs; int bn = tid / Cs;
  float mx = -3.0e38f;
  for (int kk = 0; kk < K; ++kk) mx = fmaxf(mx, Wt[((long)bn * K + kk) * Cs + c]);
  float s = 0.f;
  for (int kk = 0; kk < K; ++kk) {
    float e = expf(Wt[((long)bn * K + kk) * Cs + c] - mx);
    Wt[((long)bn * K + kk) * Cs + c] = e;
    s += e;
  }
  float inv = 1.f / s;
  for (int kk = 0; kk < K; ++kk) Wt[((long)bn * K + kk) * Cs + c] *= inv;
}

__global__ void k_maxk(const float* __restrict__ X, float* __restrict__ out,
                       int rows, int K, int C) {
  int tid = blockIdx.x * blockDim.x + threadIdx.x;
  int total = rows * C;
  if (tid >= total) return;
  int c = tid % C; int r = tid / C;
  float m = -3.0e38f;
  for (int kk = 0; kk < K; ++kk) m = fmaxf(m, X[((long)r * K + kk) * C + c]);
  out[tid] = m;
}

__global__ void k_mean1(const float* __restrict__ X, float* __restrict__ out,
                        int Bn, int Np, int C) {
  int tid = blockIdx.x * blockDim.x + threadIdx.x;
  int total = Bn * C;
  if (tid >= total) return;
  int c = tid % C; int b = tid / C;
  float s = 0.f;
  for (int i = 0; i < Np; ++i) s += X[((long)b * Np + i) * C + c];
  out[tid] = s / (float)Np;
}

// out = A + sum_{t<3} w_t * F2[b, idx[b,i,t], c], w from inverse distance
__global__ void k_interp(const float* __restrict__ F2, const int* __restrict__ idx,
                         const float* __restrict__ dist, const float* __restrict__ A,
                         float* __restrict__ out, int Bn, int N1, int N2, int C) {
  int tid = blockIdx.x * blockDim.x + threadIdx.x;
  int total = Bn * N1 * C;
  if (tid >= total) return;
  int c = tid % C; int i = (tid / C) % N1; int b = tid / (C * N1);
  long base = ((long)b * N1 + i) * 3;
  float w0 = 1.f / (dist[base + 0] + 1e-8f);
  float w1 = 1.f / (dist[base + 1] + 1e-8f);
  float w2 = 1.f / (dist[base + 2] + 1e-8f);
  float inv = 1.f / (w0 + w1 + w2);
  float acc = w0 * inv * F2[((long)b * N2 + idx[base + 0]) * C + c] +
              w1 * inv * F2[((long)b * N2 + idx[base + 1]) * C + c] +
              w2 * inv * F2[((long)b * N2 + idx[base + 2]) * C + c];
  out[tid] = A[tid] + acc;
}

// ---------------------------------------------------------------------------
// Host-side orchestration
// ---------------------------------------------------------------------------
namespace {

struct Arena {
  char* base; size_t cap; size_t off;
  void* alloc(size_t bytes) {
    size_t a = (bytes + 255) & ~(size_t)255;
    void* p = base + off; off += a; return p;
  }
  float* f(size_t n) { return (float*)alloc(n * sizeof(float)); }
  int*   i(size_t n) { return (int*)alloc(n * sizeof(int)); }
};

struct Attn { const float *wq,*bq,*wk,*bk,*wv,*bv,*wp1,*bp1,*gp,*bpn,*wp2,*bp2,
                          *gw1,*bw1,*ww1,*bwb1,*gw2,*bw2,*ww2,*bwb2; };
struct Block { const float *w1,*g1,*b1; Attn a; const float *g2,*b2,*w3,*g3,*b3; };
struct Down  { const float *w,*g,*b; };
struct UpHead{ const float *w1,*b1,*g1,*bn1,*w2,*b2; };
struct Up    { const float *w1,*b1,*g1,*bn1,*w2,*b2,*g2,*bn2; };
struct HeadP { const float *w1,*b1,*g,*bg,*w2,*b2; };

struct Cursor {
  void* const* d; int i;
  const float* operator()() { return (const float*)d[i++]; }
};

static Attn loadAttn(Cursor& c) {
  Attn a;
  a.wq=c(); a.bq=c(); a.wk=c(); a.bk=c(); a.wv=c(); a.bv=c();
  a.wp1=c(); a.bp1=c(); a.gp=c(); a.bpn=c(); a.wp2=c(); a.bp2=c();
  a.gw1=c(); a.bw1=c(); a.ww1=c(); a.bwb1=c(); a.gw2=c(); a.bw2=c();
  a.ww2=c(); a.bwb2=c();
  return a;
}
static Block loadBlock(Cursor& c) {
  Block b;
  b.w1=c(); b.g1=c(); b.b1=c(); b.a = loadAttn(c);
  b.g2=c(); b.b2=c(); b.w3=c(); b.g3=c(); b.b3=c();
  return b;
}

static void gemm(hipStream_t s, const float* X, const float* W, const float* bias,
                 float* Y, int M, int K, int N) {
  int grid = ceil_div(M, GBM) * ceil_div(N, GBN);
  k_gemm_wmma<<<grid, 256, 0, s>>>(X, W, bias, Y, M, K, N);
}

static void bn_run(hipStream_t s, Arena& ar, const float* X, const float* g,
                   const float* b, float* Y, int rows, int C, bool relu,
                   const float* resid) {
  size_t mark = ar.off;
  float* mean = ar.f(C);
  float* var  = ar.f(C);
  k_bnstats<<<C, 256, 0, s>>>(X, mean, var, rows, C);
  int total = rows * C;
  k_bn_apply<<<ceil_div(total, 256), 256, 0, s>>>(X, mean, var, g, b, resid, Y,
                                                  total, C, relu ? 1 : 0);
  ar.off = mark;  // stream-ordered reuse is safe
}

static void knn_run(hipStream_t s, const float* Q, const float* R, int* idx,
                    float* dist, int Bn, int NQ, int NR, int k) {
  dim3 grid(ceil_div(NQ, 128), Bn);
  if (k == 3)      k_knn_t<3><<<grid, 128, 0, s>>>(Q, R, idx, dist, NQ, NR);
  else if (k == 8) k_knn_t<8><<<grid, 128, 0, s>>>(Q, R, idx, dist, NQ, NR);
  else             k_knn_t<16><<<grid, 128, 0, s>>>(Q, R, idx, dist, NQ, NR);
}

static void pt_layer(hipStream_t s, Arena& ar, const float* p, const float* x,
                     const Attn& A, int Bn, int Np, int C, int K, float* out) {
  size_t mark = ar.off;
  const int rows = Bn * Np;
  const int cs = C / 8;
  const int rk = rows * K;

  float* xq = ar.f((size_t)rows * C);
  float* xk = ar.f((size_t)rows * C);
  float* xv = ar.f((size_t)rows * C);
  gemm(s, x, A.wq, A.bq, xq, rows, C, C);
  gemm(s, x, A.wk, A.bk, xk, rows, C, C);
  gemm(s, x, A.wv, A.bv, xv, rows, C, C);

  int* idx = ar.i((size_t)rows * K);
  float* dist = ar.f((size_t)rows * K);
  knn_run(s, p, p, idx, dist, Bn, Np, Np, K);

  float* pr = ar.f((size_t)rk * 3);
  k_gather_sub3<<<ceil_div(rk * 3, 256), 256, 0, s>>>(p, idx, p, pr, Bn, Np, K, Np);

  float* pe1 = ar.f((size_t)rk * 3);
  gemm(s, pr, A.wp1, A.bp1, pe1, rk, 3, 3);
  bn_run(s, ar, pe1, A.gp, A.bpn, pe1, rk, 3, true, nullptr);
  float* pe = ar.f((size_t)rk * C);
  gemm(s, pe1, A.wp2, A.bp2, pe, rk, 3, C);

  // fused gather: w = xk[idx] - xq + pe
  float* w = ar.f((size_t)rk * C);
  k_qkpe_g<<<ceil_div(rk * C, 256), 256, 0, s>>>(xk, idx, xq, pe, w, Bn, Np, K, C);
  bn_run(s, ar, w, A.gw1, A.bw1, w, rk, C, true, nullptr);

  float* w2 = ar.f((size_t)rk * cs);
  gemm(s, w, A.ww1, A.bwb1, w2, rk, C, cs);
  bn_run(s, ar, w2, A.gw2, A.bw2, w2, rk, cs, true, nullptr);

  float* w3 = ar.f((size_t)rk * cs);
  gemm(s, w2, A.ww2, A.bwb2, w3, rk, cs, cs);
  k_softmax_k<<<ceil_div(rows * cs, 256), 256, 0, s>>>(w3, rows, K, cs);

  // fused gather: out = sum_k (xv[idx] + pe) * softmax-w
  k_attn_agg_g<<<ceil_div(rows * C, 256), 256, 0, s>>>(xv, idx, pe, w3, out,
                                                       Bn, Np, K, C, cs);
  ar.off = mark;
}

static void pt_block(hipStream_t s, Arena& ar, const float* p, float* x,
                     const Block& blk, int Bn, int Np, int C, int K) {
  size_t mark = ar.off;
  const int rows = Bn * Np;
  float* y = ar.f((size_t)rows * C);
  gemm(s, x, blk.w1, nullptr, y, rows, C, C);
  bn_run(s, ar, y, blk.g1, blk.b1, y, rows, C, true, nullptr);

  float* y2 = ar.f((size_t)rows * C);
  pt_layer(s, ar, p, y, blk.a, Bn, Np, C, K, y2);
  bn_run(s, ar, y2, blk.g2, blk.b2, y2, rows, C, true, nullptr);

  float* y3 = ar.f((size_t)rows * C);
  gemm(s, y2, blk.w3, nullptr, y3, rows, C, C);
  bn_run(s, ar, y3, blk.g3, blk.b3, x, rows, C, true, x);  // relu(bn(.)+x) -> x
  ar.off = mark;
}

static void tdown(hipStream_t s, Arena& ar, const float* p, const float* x,
                  const Down& D, int Bn, int Np, int stride, int K, int Cin,
                  int Cout, float* new_p, float* out) {
  size_t mark = ar.off;
  const int m = Np / stride;
  int* sidx = ar.i((size_t)Bn * m);
  k_fps<<<Bn, FPS_THREADS, 0, s>>>(p, sidx, Np, m);
  k_gather_pts<<<ceil_div(Bn * m * 3, 256), 256, 0, s>>>(p, sidx, new_p, Bn, m, Np);

  int* idx = ar.i((size_t)Bn * m * K);
  float* dist = ar.f((size_t)Bn * m * K);
  knn_run(s, new_p, p, idx, dist, Bn, m, Np, K);

  const int rk = Bn * m * K;
  float* rel = ar.f((size_t)rk * 3);
  k_gather_sub3<<<ceil_div(rk * 3, 256), 256, 0, s>>>(p, idx, new_p, rel, Bn, m, K, Np);
  float* gf = ar.f((size_t)rk * Cin);
  k_gather_feat<<<ceil_div(rk * Cin, 256), 256, 0, s>>>(x, idx, gf, Bn, m, K, Cin, Np);
  float* gx = ar.f((size_t)rk * (3 + Cin));
  k_concat<<<ceil_div(rk * (3 + Cin), 256), 256, 0, s>>>(rel, gf, gx, rk, 3, Cin);

  float* y = ar.f((size_t)rk * Cout);
  gemm(s, gx, D.w, nullptr, y, rk, 3 + Cin, Cout);
  bn_run(s, ar, y, D.g, D.b, y, rk, Cout, true, nullptr);
  k_maxk<<<ceil_div(Bn * m * Cout, 256), 256, 0, s>>>(y, out, Bn * m, K, Cout);
  ar.off = mark;
}

static void tup_head(hipStream_t s, Arena& ar, const float* x, const UpHead& U,
                     int Bn, int Np, int C, float* out) {
  size_t mark = ar.off;
  float* mn = ar.f((size_t)Bn * C);
  k_mean1<<<ceil_div(Bn * C, 256), 256, 0, s>>>(x, mn, Bn, Np, C);
  float* g = ar.f((size_t)Bn * C);
  gemm(s, mn, U.w2, U.b2, g, Bn, C, C);
  k_relu<<<ceil_div(Bn * C, 256), 256, 0, s>>>(g, Bn * C);
  float* xc = ar.f((size_t)Bn * Np * 2 * C);
  k_concat_bcast<<<ceil_div(Bn * Np * 2 * C, 256), 256, 0, s>>>(x, g, xc, Bn, Np, C);
  gemm(s, xc, U.w1, U.b1, out, Bn * Np, 2 * C, C);
  bn_run(s, ar, out, U.g1, U.bn1, out, Bn * Np, C, true, nullptr);
  ar.off = mark;
}

static void tup(hipStream_t s, Arena& ar, const float* p1, const float* x1,
                const float* p2, const float* x2, const Up& U, int Bn, int N1,
                int N2, int Cout, int Cin, float* out) {
  size_t mark = ar.off;
  float* a = ar.f((size_t)Bn * N1 * Cout);
  gemm(s, x1, U.w1, U.b1, a, Bn * N1, Cout, Cout);
  bn_run(s, ar, a, U.g1, U.bn1, a, Bn * N1, Cout, true, nullptr);
  float* bb = ar.f((size_t)Bn * N2 * Cout);
  gemm(s, x2, U.w2, U.b2, bb, Bn * N2, Cin, Cout);
  bn_run(s, ar, bb, U.g2, U.bn2, bb, Bn * N2, Cout, true, nullptr);
  int* idx = ar.i((size_t)Bn * N1 * 3);
  float* dist = ar.f((size_t)Bn * N1 * 3);
  knn_run(s, p1, p2, idx, dist, Bn, N1, N2, 3);
  k_interp<<<ceil_div(Bn * N1 * Cout, 256), 256, 0, s>>>(bb, idx, dist, a, out,
                                                         Bn, N1, N2, Cout);
  ar.off = mark;
}

static void head_run(hipStream_t s, Arena& ar, const float* x, const HeadP& H,
                     int Bn, int Np, int Cin, int Cout, float* outT) {
  size_t mark = ar.off;
  float* y = ar.f((size_t)Bn * Np * Cin);
  gemm(s, x, H.w1, H.b1, y, Bn * Np, Cin, Cin);
  bn_run(s, ar, y, H.g, H.bg, y, Bn * Np, Cin, true, nullptr);
  float* z = ar.f((size_t)Bn * Np * Cout);
  gemm(s, y, H.w2, H.b2, z, Bn * Np, Cin, Cout);
  k_transpose_out<<<ceil_div(Bn * Np * Cout, 256), 256, 0, s>>>(z, outT, Bn, Np, Cout);
  ar.off = mark;
}

} // namespace

extern "C" void kernel_launch(void* const* d_in, const int* in_sizes, int n_in,
                              void* d_out, int out_size, void* d_ws, size_t ws_size,
                              hipStream_t stream) {
  (void)in_sizes; (void)n_in; (void)out_size;
  const int Bn = 2;
  const int Ns[5] = {8192, 2048, 512, 128, 32};
  const int Cs[5] = {32, 64, 128, 256, 512};
  const int Ks[5] = {8, 16, 16, 16, 16};
  const int IN_CH = 6;

  const float* data = (const float*)d_in[0];
  Cursor cur{d_in, 1};

  // encoder params (dict order: down{w,g,b}, block)
  Down  encD[5]; Block encB[5];
  for (int i = 0; i < 5; ++i) {
    encD[i].w = cur(); encD[i].g = cur(); encD[i].b = cur();
    encB[i] = loadBlock(cur);
  }
  // dec5: up{w1,b1,g1,bn1,w2,b2}, block
  UpHead dec5U;
  dec5U.w1 = cur(); dec5U.b1 = cur(); dec5U.g1 = cur(); dec5U.bn1 = cur();
  dec5U.w2 = cur(); dec5U.b2 = cur();
  Block dec5B = loadBlock(cur);
  // dec4..dec1 in insertion order (i = 3,2,1,0)
  Up decU[4]; Block decB[4];
  for (int j = 0; j < 4; ++j) {
    decU[j].w1 = cur(); decU[j].b1 = cur(); decU[j].g1 = cur(); decU[j].bn1 = cur();
    decU[j].w2 = cur(); decU[j].b2 = cur(); decU[j].g2 = cur(); decU[j].bn2 = cur();
    decB[j] = loadBlock(cur);
  }
  HeadP clsH, edgeH;
  clsH.w1 = cur(); clsH.b1 = cur(); clsH.g = cur(); clsH.bg = cur();
  clsH.w2 = cur(); clsH.b2 = cur();
  edgeH.w1 = cur(); edgeH.b1 = cur(); edgeH.g = cur(); edgeH.bg = cur();
  edgeH.w2 = cur(); edgeH.b2 = cur();

  Arena ar{(char*)d_ws, ws_size, 0};

  // ------------- encoder level 0 -------------
  float* p0 = ar.f((size_t)Bn * Ns[0] * 3);
  float* x0 = ar.f((size_t)Bn * Ns[0] * IN_CH);
  k_transpose_in<<<ceil_div(Bn * Ns[0] * IN_CH, 256), 256, 0, stream>>>(
      data, x0, p0, Bn, IN_CH, Ns[0]);

  float* ps[5]; float* xs[5];
  ps[0] = p0;
  xs[0] = ar.f((size_t)Bn * Ns[0] * Cs[0]);
  gemm(stream, x0, encD[0].w, nullptr, xs[0], Bn * Ns[0], IN_CH, Cs[0]);
  bn_run(stream, ar, xs[0], encD[0].g, encD[0].b, xs[0], Bn * Ns[0], Cs[0], true, nullptr);
  pt_block(stream, ar, ps[0], xs[0], encB[0], Bn, Ns[0], Cs[0], Ks[0]);

  // ------------- encoder levels 1..4 -------------
  for (int i = 1; i < 5; ++i) {
    ps[i] = ar.f((size_t)Bn * Ns[i] * 3);
    xs[i] = ar.f((size_t)Bn * Ns[i] * Cs[i]);
    tdown(stream, ar, ps[i - 1], xs[i - 1], encD[i], Bn, Ns[i - 1],
          Ns[i - 1] / Ns[i], Ks[i], Cs[i - 1], Cs[i], ps[i], xs[i]);
    pt_block(stream, ar, ps[i], xs[i], encB[i], Bn, Ns[i], Cs[i], Ks[i]);
  }

  // ------------- decoder -------------
  float* x5 = ar.f((size_t)Bn * Ns[4] * Cs[4]);
  tup_head(stream, ar, xs[4], dec5U, Bn, Ns[4], Cs[4], x5);
  pt_block(stream, ar, ps[4], x5, dec5B, Bn, Ns[4], Cs[4], Ks[4]);
  xs[4] = x5;

  for (int i = 3; i >= 0; --i) {
    int j = 3 - i;  // decU/decB loaded in order dec4,dec3,dec2,dec1
    float* xn = ar.f((size_t)Bn * Ns[i] * Cs[i]);
    tup(stream, ar, ps[i], xs[i], ps[i + 1], xs[i + 1], decU[j], Bn, Ns[i],
        Ns[i + 1], Cs[i], Cs[i + 1], xn);
    pt_block(stream, ar, ps[i], xn, decB[j], Bn, Ns[i], Cs[i], Ks[i]);
    xs[i] = xn;
  }

  // ------------- heads -------------
  float* out = (float*)d_out;
  head_run(stream, ar, xs[0], clsH, Bn, Ns[0], 32, 13, out);
  head_run(stream, ar, xs[0], edgeH, Bn, Ns[0], 32, 2, out + (size_t)Bn * 13 * Ns[0]);
}
